// Encoder_58394375356777
// MI455X (gfx1250) — matrix operand order repaired and checked by reading, not compiled, not run on hardware
//
#include <hip/hip_runtime.h>

typedef __bf16 bf16_t;
typedef __bf16 v16bf __attribute__((ext_vector_type(16)));
typedef __bf16 v8bf  __attribute__((ext_vector_type(8)));
typedef float  v8f   __attribute__((ext_vector_type(8)));

// Exact pointee type expected by the async-to-LDS builtin (from compiler diagnostic):
// '__attribute__((__vector_size__(4 * sizeof(int)))) int'
typedef int vs4i __attribute__((vector_size(16)));
typedef __attribute__((address_space(1))) vs4i gvs4i;   // global (cuda_device) AS
typedef __attribute__((address_space(3))) vs4i lvs4i;   // LDS (cuda_shared) AS

#define B_   64
#define T_   512
#define H_   512
#define L_   4
#define K_   1024      // concat(inp_t, h) reduction depth
#define NWG  16        // persistent workgroups (grid barrier participants)
#define WGSL 32        // h-columns owned per WG
#define SLICE 128      // z-columns per WG (4 gates * WGSL)
#define NTHR 256       // 8 wave32 waves

#if __has_builtin(__builtin_amdgcn_global_load_async_to_lds_b128)
#define HAVE_ASYNC_LDS 1
#else
#define HAVE_ASYNC_LDS 0
#endif

__device__ __forceinline__ void wait_asynccnt0() {
#if __has_builtin(__builtin_amdgcn_s_wait_asynccnt)
  __builtin_amdgcn_s_wait_asynccnt(0);
#else
  asm volatile("s_wait_asynccnt 0x0" ::: "memory");
#endif
}

// ---------------- weight pack: WUt[l][n][k] = bf16( k<512 ? W[l][k][n] : U[l][k-512][n] )
__global__ void pack_wu_kernel(const float* __restrict__ W, const float* __restrict__ U,
                               bf16_t* __restrict__ WUt) {
  size_t i = (size_t)blockIdx.x * NTHR + threadIdx.x;   // over L*2048*1024
  int k = (int)(i & 1023);
  size_t nn = i >> 10;
  int n = (int)(nn & 2047);
  int l = (int)(nn >> 11);
  float v = (k < 512) ? W[(size_t)l * H_ * 2048 + (size_t)k * 2048 + n]
                      : U[(size_t)l * H_ * 2048 + (size_t)(k - 512) * 2048 + n];
  WUt[i] = (bf16_t)v;
}

// ---------------- input pack: a0[(t*B+b)*H + k] = bf16(x[b][t][k])
__global__ void pack_x_kernel(const float* __restrict__ x, bf16_t* __restrict__ a0) {
  size_t i = (size_t)blockIdx.x * NTHR + threadIdx.x;   // over T*B*H
  int k = (int)(i & 511);
  size_t tb = i >> 9;
  int b = (int)(tb & 63);
  int t = (int)(tb >> 6);
  a0[i] = (bf16_t)x[(size_t)b * T_ * H_ + (size_t)t * H_ + k];
}

__device__ __forceinline__ void grid_barrier(unsigned* cnt) {
  __syncthreads();
  if (threadIdx.x == 0) {
    __threadfence();
    unsigned arrive = atomicAdd(cnt, 1u) + 1u;
    unsigned target = ((arrive + NWG - 1u) / NWG) * NWG;
    while (__hip_atomic_load(cnt, __ATOMIC_RELAXED, __HIP_MEMORY_SCOPE_AGENT) < target) {
      __builtin_amdgcn_s_sleep(1);
    }
    __threadfence();
  }
  __syncthreads();
}

__device__ __forceinline__ float sigf(float v) { return 1.0f / (1.0f + __expf(-v)); }

// ---------------- persistent per-layer LSTM scan
__launch_bounds__(NTHR, 1)
__global__ void lstm_scan_kernel(const bf16_t* __restrict__ inpA,     // [T*B][H] bf16 layer input
                                 const float*  __restrict__ resPrev,  // [T*B][H] fp32, null on layer0
                                 const bf16_t* __restrict__ WUt,      // [2048][1024] bf16 (this layer)
                                 const float*  __restrict__ bias,     // [2048] fp32 (this layer)
                                 float*        __restrict__ outF,     // fp32 layer output (+residual)
                                 bf16_t*       __restrict__ inpAnext, // bf16 next-layer input, null on L3
                                 bf16_t*       __restrict__ hbuf,     // [2][B*H] bf16 h exchange
                                 float*        __restrict__ statesOut,// d_out states region, layer0 only
                                 unsigned*     __restrict__ syncCnt,
                                 int finalLayout) {                   // 0: [t*B+b][H], 1: [b][T][H]
  extern __shared__ char smem[];
  bf16_t* a_s = (bf16_t*)smem;            // [B_][K_] bf16 = 128KB staging
  float*  z_s = (float*)smem;             // alias: [B_][SLICE] fp32 = 32KB
  __shared__ float c_s[B_ * WGSL];        // cell state, resident across all T steps

  const int tid  = threadIdx.x;
  const int wg   = blockIdx.x;
  const int lane = tid & 31;
  const int wv   = tid >> 5;

  // init: c = 0, h(t=-1) = 0 (hbuf[0]) cooperatively across the grid
  for (int i = tid; i < B_ * WGSL; i += NTHR) c_s[i] = 0.0f;
  {
    unsigned* h0 = (unsigned*)hbuf;       // B_*H_ bf16 = 16384 dwords
    for (int i = wg * NTHR + tid; i < B_ * H_ / 2; i += NWG * NTHR) h0[i] = 0u;
  }
  __threadfence();
  grid_barrier(syncCnt);

  // wave-constant GEMM geometry
  const int s0    = wv * 16;                          // slice-col base of this wave's n-tile
  const int gate0 = s0 >> 5;                          // which gate this wave computes
  const int nbase = gate0 * H_ + wg * WGSL + (s0 & 31);
  const bf16_t* bptr = WUt + (size_t)(nbase + (lane & 15)) * K_ + ((lane >> 4) << 4);
  const int am  = lane & 15;                          // A row within m-tile
  const int ahi = (lane >> 4) << 4;                   // +16 K-offset for lanes 16..31

  for (int t = 0; t < T_; ++t) {
    const int rp = t & 1;
    const bf16_t* hread  = hbuf + (size_t)rp * (B_ * H_);
    bf16_t*       hwrite = hbuf + (size_t)(rp ^ 1) * (B_ * H_);

    // ---- stage A = [inp_t | h_{t-1}] into LDS (16B chunks) ----
    const bf16_t* xrow = inpA + (size_t)t * B_ * H_;
#if HAVE_ASYNC_LDS
    // CDNA5 async direct-to-LDS copy: one instruction per 16B, tracked by ASYNCcnt,
    // no VGPR round-trip.
    for (int ch = tid; ch < (B_ * K_ / 8); ch += NTHR) {
      int row = ch >> 7;
      int wi  = ch & 127;
      const bf16_t* src = (wi < 64) ? (xrow  + (size_t)row * H_ + wi * 8)
                                    : (hread + (size_t)row * H_ + (wi - 64) * 8);
      __builtin_amdgcn_global_load_async_to_lds_b128(
          (gvs4i*)src, (lvs4i*)(a_s + (size_t)row * K_ + wi * 8), 0, 0);
    }
    wait_asynccnt0();
#else
    for (int ch = tid; ch < (B_ * K_ / 8); ch += NTHR) {
      int row = ch >> 7;
      int wi  = ch & 127;
      v8bf v = (wi < 64) ? *(const v8bf*)(xrow  + (size_t)row * H_ + wi * 8)
                         : *(const v8bf*)(hread + (size_t)row * H_ + (wi - 64) * 8);
      *(v8bf*)(a_s + (size_t)row * K_ + wi * 8) = v;
    }
#endif
    __syncthreads();

    // ---- z[64 x SLICE] = A @ WUt_slice : 32 k-tiles x 4 m-tiles of v_wmma_f32_16x16x32_bf16
    v8f acc[4];
#pragma unroll
    for (int mt = 0; mt < 4; ++mt) acc[mt] = (v8f)0.0f;

    for (int kt = 0; kt < K_ / 32; ++kt) {
      v16bf bf = *(const v16bf*)(bptr + kt * 32);     // contiguous 32B per lane
#pragma unroll
      for (int mt = 0; mt < 4; ++mt) {
        const bf16_t* ap = a_s + (size_t)(mt * 16 + am) * K_ + kt * 32 + ahi;
        v8bf lo = *(const v8bf*)ap;                   // K = base .. base+7
        v8bf hi = *(const v8bf*)(ap + 8);             // K = base+8 .. base+15
        v16bf a = __builtin_shufflevector(lo, hi, 0, 1, 2, 3, 4, 5, 6, 7,
                                                  8, 9, 10, 11, 12, 13, 14, 15);
        acc[mt] = __builtin_amdgcn_wmma_f32_16x16x32_bf16(
            false, a, false, bf, (short)0, acc[mt], false, false);
      }
    }
    __syncthreads();   // all waves done reading a_s before aliasing z_s over it

    // ---- scatter D tiles to LDS: VGPR r -> M = r + 8*(lane>=16), N = lane&15
#pragma unroll
    for (int mt = 0; mt < 4; ++mt) {
#pragma unroll
      for (int r = 0; r < 8; ++r) {
        int M = mt * 16 + r + ((lane >> 4) << 3);
        z_s[M * SLICE + s0 + (lane & 15)] = acc[mt][r];
      }
    }
    __syncthreads();

    // ---- gates + cell update + outputs (8 elements/thread) ----
    for (int e = tid; e < B_ * WGSL; e += NTHR) {
      int b  = e >> 5;
      int jj = e & 31;
      int jglob = wg * WGSL + jj;
      float zi = z_s[b * SLICE + jj]      + bias[0 * H_ + jglob];
      float zf = z_s[b * SLICE + 32 + jj] + bias[1 * H_ + jglob];
      float zg = z_s[b * SLICE + 64 + jj] + bias[2 * H_ + jglob];
      float zo = z_s[b * SLICE + 96 + jj] + bias[3 * H_ + jglob];
      float c  = sigf(zf) * c_s[e] + sigf(zi) * tanhf(zg);
      c_s[e] = c;
      float h = sigf(zo) * tanhf(c);
      hwrite[b * H_ + jglob] = (bf16_t)h;
      float resv = resPrev ? (h + resPrev[((size_t)t * B_ + b) * H_ + jglob]) : h;
      if (finalLayout) outF[(size_t)b * T_ * H_ + (size_t)t * H_ + jglob] = resv;
      else             outF[((size_t)t * B_ + b) * H_ + jglob] = resv;
      if (inpAnext) inpAnext[((size_t)t * B_ + b) * H_ + jglob] = (bf16_t)resv;
      if (statesOut && t == T_ - 1) {
#pragma unroll
        for (int rep = 0; rep < L_; ++rep)
          statesOut[(size_t)rep * B_ * H_ + (size_t)b * H_ + jglob] = h;
      }
    }
    __threadfence();
    grid_barrier(syncCnt);   // h_t visible to all WGs before step t+1
  }
}

extern "C" void kernel_launch(void* const* d_in, const int* in_sizes, int n_in,
                              void* d_out, int out_size, void* d_ws, size_t ws_size,
                              hipStream_t stream) {
  (void)in_sizes; (void)n_in; (void)out_size; (void)ws_size;
  const float* x  = (const float*)d_in[0];
  const float* W  = (const float*)d_in[1];
  const float* U  = (const float*)d_in[2];
  const float* bv = (const float*)d_in[3];
  float* out = (float*)d_out;                       // [B*T*H] rnn_output, then [L*B*H] states
  float* states = out + (size_t)B_ * T_ * H_;

  // workspace carve-up
  char* ws = (char*)d_ws;
  size_t off = 0;
  bf16_t* WUt = (bf16_t*)(ws + off); off += (size_t)L_ * 2048 * K_ * sizeof(bf16_t);   // 16 MB
  bf16_t* a0  = (bf16_t*)(ws + off); off += (size_t)T_ * B_ * H_ * sizeof(bf16_t);     // 32 MB
  bf16_t* a1  = (bf16_t*)(ws + off); off += (size_t)T_ * B_ * H_ * sizeof(bf16_t);     // 32 MB
  float*  r0  = (float*)(ws + off);  off += (size_t)T_ * B_ * H_ * sizeof(float);      // 64 MB
  float*  r1  = (float*)(ws + off);  off += (size_t)T_ * B_ * H_ * sizeof(float);      // 64 MB
  bf16_t* hb  = (bf16_t*)(ws + off); off += (size_t)2 * B_ * H_ * sizeof(bf16_t);      // 128 KB
  unsigned* cnt = (unsigned*)(ws + off);

  (void)hipMemsetAsync(cnt, 0, sizeof(unsigned), stream);

  pack_wu_kernel<<<(L_ * 2048 * K_) / NTHR, NTHR, 0, stream>>>(W, U, WUt);
  pack_x_kernel<<<(T_ * B_ * H_) / NTHR, NTHR, 0, stream>>>(x, a0);

  const size_t shmem = (size_t)B_ * K_ * sizeof(bf16_t);   // 128 KB dynamic LDS
  (void)hipFuncSetAttribute(reinterpret_cast<const void*>(lstm_scan_kernel),
                            hipFuncAttributeMaxDynamicSharedMemorySize, (int)shmem);

  const size_t wstride = (size_t)2048 * K_;
  // layer 0: input a0, no residual, writes r0 + a1, final h -> states (x4)
  lstm_scan_kernel<<<NWG, NTHR, shmem, stream>>>(a0, nullptr, WUt + 0 * wstride,
                                                 bv + 0 * 2048, r0, a1, hb, states, cnt, 0);
  // layer 1: input a1, residual r0, writes r1 + a0
  lstm_scan_kernel<<<NWG, NTHR, shmem, stream>>>(a1, r0, WUt + 1 * wstride,
                                                 bv + 1 * 2048, r1, a0, hb, nullptr, cnt, 0);
  // layer 2: input a0, residual r1, writes r0 + a1
  lstm_scan_kernel<<<NWG, NTHR, shmem, stream>>>(a0, r1, WUt + 2 * wstride,
                                                 bv + 2 * 2048, r0, a1, hb, nullptr, cnt, 0);
  // layer 3: input a1, residual r0, writes rnn_output directly as [B][T][H]
  lstm_scan_kernel<<<NWG, NTHR, shmem, stream>>>(a1, r0, WUt + 3 * wstride,
                                                 bv + 3 * 2048, out, nullptr, hb, nullptr, cnt, 1);
}
